// Detect_30262339568078
// MI455X (gfx1250) — matrix operand order repaired and checked by reading, not compiled, not run on hardware
//
#include <hip/hip_runtime.h>
#include <hip/hip_bf16.h>

typedef __attribute__((ext_vector_type(16))) _Float16 v16h;
typedef __attribute__((ext_vector_type(8)))  _Float16 v8h;
typedef __attribute__((ext_vector_type(8)))  float    v8f;

#define NA_ 3
#define NO_ 85

// ---------------------------------------------------------------------------
// Head GEMM: out[b,a,y,x,o] = sum_k X[b,k,m]*W[a*85+o,k] + bias[a*85+o]
// WMMA tile: M = 16 spatial positions, N = 16 output channels, K = 32 (f16).
// 8 waves per block; each wave owns 2 of the 16 output-channel tiles (255->256
// padded).  X tile staged once per block into LDS as f16 (layout [n][c], row
// padded by 8 halves -> 16B-aligned, bank-conflict-free A-fragment reads).
// ---------------------------------------------------------------------------
template<int C>
__global__ __launch_bounds__(256) void head_wmma_kernel(
    const float* __restrict__ x,    // [8, C, nyx]
    const float* __restrict__ w,    // [255, C]
    const float* __restrict__ bias, // [255]
    float* __restrict__ out,        // [8, 3, nyx, 85]
    int nyx)
{
  constexpr int PADC = C + 8;
  __shared__ __align__(16) _Float16 lds_x[16 * PADC];
  const int b  = blockIdx.y;
  const int m0 = blockIdx.x * 16;
  const int t  = threadIdx.x;

  // Stage X tile (16 spatial x C channels) -> LDS f16, layout lds_x[n][c].
  {
    const float* xb = x + ((size_t)b * C) * nyx + m0;
    const int n = t & 15;
    const int cbase = t >> 4;
#pragma unroll
    for (int i = 0; i < C / 16; ++i) {
      const int c = cbase + i * 16;
      lds_x[n * PADC + c] = (_Float16)xb[(size_t)c * nyx + n];
    }
  }
  __syncthreads();

  const int wave = t >> 5;     // wave32: 8 waves per 256-thread block
  const int lane = t & 31;
  const int kg   = lane >> 4;  // lane group 0/1
  const int ln   = lane & 15;

  const int o0 = (wave * 2 + 0) * 16 + ln;   // output channel for B-tile 0
  const int o1 = (wave * 2 + 1) * 16 + ln;   // output channel for B-tile 1
  const float* w0r = w + (size_t)(o0 < 254 ? o0 : 254) * C;
  const float* w1r = w + (size_t)(o1 < 254 ? o1 : 254) * C;

  v8f acc0 = {};
  v8f acc1 = {};

  for (int k0 = 0; k0 < C; k0 += 32) {
    // A fragment (ISA 16-bit A 16x32): lane row m=ln; elements 0..7 hold
    // K = k0+kg*8+e, elements 8..15 hold K = k0+16+kg*8+(e-8).
    v16h a;
    {
      const _Float16* r = &lds_x[ln * PADC + k0 + kg * 8];
      v8h alo = *(const v8h*)(r);
      v8h ahi = *(const v8h*)(r + 16);
#pragma unroll
      for (int e = 0; e < 8; ++e) { a[e] = alo[e]; a[e + 8] = ahi[e]; }
    }
    // B fragments (K x 16 o-channels): lane col n=ln, element e -> K = k0+kg*8+e (e<8), k0+16+kg*8+(e-8) (e>=8).
    v16h bf0, bf1;
    {
      const float* p0 = w0r + k0 + kg * 8;
      const float* p1 = w1r + k0 + kg * 8;
#pragma unroll
      for (int e = 0; e < 16; ++e) {
        bf0[e] = (_Float16)p0[e + (e >= 8 ? 8 : 0)];
        bf1[e] = (_Float16)p1[e + (e >= 8 ? 8 : 0)];
      }
    }
    acc0 = __builtin_amdgcn_wmma_f32_16x16x32_f16(false, a, false, bf0,
                                                  (short)0, acc0, false, false);
    acc1 = __builtin_amdgcn_wmma_f32_16x16x32_f16(false, a, false, bf1,
                                                  (short)0, acc1, false, false);
  }

  // D layout: element j -> spatial row m = kg*8 + j, column = ln.
  const int mbase = m0 + kg * 8;
#pragma unroll
  for (int tile = 0; tile < 2; ++tile) {
    const int of = tile ? o1 : o0;
    const v8f acc = tile ? acc1 : acc0;
    if (of < 255) {
      const float bv = bias[of];
      const int ai = of / NO_;
      const int oo = of % NO_;
      float* ob = out + (((size_t)(b * NA_ + ai)) * nyx + mbase) * NO_ + oo;
#pragma unroll
      for (int j = 0; j < 8; ++j) ob[(size_t)j * NO_] = acc[j] + bv;
    }
  }
}

// ---------------------------------------------------------------------------
// Box-average pooling over jax.image.resize'd features, done analytically:
// resize is separable => box sum = sum_j Wy[j] * sum_i Wx[i] * X[b,c,j,i]
// with Wx[i] = sum over box output columns of the (column-normalized) resize
// weight.  Matches jax half-pixel sampling + Keys cubic (a=-0.5) / triangle.
// One block per (box, level); threads own channels.
// ---------------------------------------------------------------------------
__global__ __launch_bounds__(256) void obj_pool_kernel(
    const float* __restrict__ x0, const float* __restrict__ x1,
    const float* __restrict__ x2, const float* __restrict__ labels,
    float* __restrict__ obj_out,  // [64, 896]
    float* __restrict__ gt_out)   // [64]
{
  const int n   = blockIdx.x;   // box 0..63
  const int lev = blockIdx.y;   // 0..2
  const int t   = threadIdx.x;

  const int   insz  = (lev == 0) ? 80 : ((lev == 1) ? 40 : 20);
  const int   C     = (lev == 0) ? 128 : ((lev == 1) ? 256 : 512);
  const int   coff  = (lev == 0) ? 0 : ((lev == 1) ? 128 : 384);
  const float* xp   = (lev == 0) ? x0 : ((lev == 1) ? x1 : x2);
  const bool  cubic = (lev != 0);

  __shared__ float Wx[80], Wy[80];
  __shared__ int bx1, bx2, by1, by2, bimg;
  __shared__ float area;
  __shared__ int ixlo, ixhi, iylo, iyhi;

  if (t == 0) {
    const float xc = labels[n * 6 + 2], yc = labels[n * 6 + 3];
    const float bw = labels[n * 6 + 4], bh = labels[n * 6 + 5];
    int X1 = (int)rintf((xc - 0.5f * bw) * 100.0f);
    int X2 = (int)rintf((xc + 0.5f * bw) * 100.0f);
    int Y1 = (int)rintf((yc - 0.5f * bh) * 100.0f);
    int Y2 = (int)rintf((yc + 0.5f * bh) * 100.0f);
    X1 = min(max(X1, 0), 100); X2 = min(max(X2, 0), 100);
    Y1 = min(max(Y1, 0), 100); Y2 = min(max(Y2, 0), 100);
    if (X2 == X1) { if (X2 < 100) X2++; else X1--; }
    if (Y2 == Y1) { if (Y2 < 100) Y2++; else Y1--; }
    bx1 = X1; bx2 = X2; by1 = Y1; by2 = Y2;
    bimg = (int)labels[n * 6 + 0];
    area = (float)((Y2 - Y1) * (X2 - X1));
    ixlo = 1 << 30; ixhi = -1; iylo = 1 << 30; iyhi = -1;
    if (lev == 0) gt_out[n] = labels[n * 6 + 1];
  }
  if (t < insz) { Wx[t] = 0.0f; Wy[t] = 0.0f; }
  __syncthreads();

  // threads 0..99 handle x output columns; 100..199 handle y output columns
  if (t < 200) {
    const bool isx = (t < 100);
    const int  col = isx ? t : (t - 100);
    const int  lo_ = isx ? bx1 : by1;
    const int  hi_ = isx ? bx2 : by2;
    if (col >= lo_ && col < hi_) {
      const float sf = (col + 0.5f) * ((float)insz / 100.0f) - 0.5f;
      const int i0 = (int)floorf(sf);
      const int tl = cubic ? i0 - 1 : i0;
      const int th = cubic ? i0 + 2 : i0 + 1;
      float tw[4]; int ti[4]; int nt = 0; float wsum = 0.0f;
      for (int i = tl; i <= th; ++i) {
        if (i < 0 || i >= insz) continue;
        const float d = fabsf(sf - (float)i);
        float kv;
        if (cubic) {  // Keys cubic, a = -0.5 (jax _fill_keys_cubic_kernel)
          kv = (d >= 2.0f) ? 0.0f
             : (d >= 1.0f) ? (((-0.5f * d + 2.5f) * d - 4.0f) * d + 2.0f)
                           : (((1.5f * d - 2.5f) * d) * d + 1.0f);
        } else {      // triangle
          kv = fmaxf(0.0f, 1.0f - d);
        }
        ti[nt] = i; tw[nt] = kv; wsum += kv; ++nt;
      }
      const float inv = 1.0f / wsum;  // jax column-sum normalization
      float* W = isx ? Wx : Wy;
      for (int k = 0; k < nt; ++k) atomicAdd(&W[ti[k]], tw[k] * inv);
      if (isx) { atomicMin(&ixlo, ti[0]); atomicMax(&ixhi, ti[nt - 1]); }
      else     { atomicMin(&iylo, ti[0]); atomicMax(&iyhi, ti[nt - 1]); }
    }
  }
  __syncthreads();

  const int j0 = iylo, j1 = iyhi, i0_ = ixlo, i1_ = ixhi;
  const float ia = 1.0f / area;
  for (int c = t; c < C; c += 256) {
    const float* plane = xp + ((size_t)bimg * C + c) * (size_t)(insz * insz);
    float s = 0.0f;
    for (int j = j0; j <= j1; ++j) {
      const float wy = Wy[j];
      const float* rowp = plane + j * insz;
      float rs = 0.0f;
      for (int i = i0_; i <= i1_; ++i) rs += Wx[i] * rowp[i];
      s += wy * rs;
    }
    obj_out[n * 896 + coff + c] = s * ia;
  }
}

// ---------------------------------------------------------------------------
extern "C" void kernel_launch(void* const* d_in, const int* in_sizes, int n_in,
                              void* d_out, int out_size, void* d_ws, size_t ws_size,
                              hipStream_t stream) {
  (void)in_sizes; (void)n_in; (void)out_size; (void)d_ws; (void)ws_size;
  const float* x0     = (const float*)d_in[0];
  const float* x1     = (const float*)d_in[1];
  const float* x2     = (const float*)d_in[2];
  const float* labels = (const float*)d_in[3];
  const float* w0     = (const float*)d_in[4];
  const float* b0     = (const float*)d_in[5];
  const float* w1     = (const float*)d_in[6];
  const float* b1     = (const float*)d_in[7];
  const float* w2     = (const float*)d_in[8];
  const float* b2     = (const float*)d_in[9];

  float* out = (float*)d_out;
  float* p0  = out;                      // 8*3*80*80*85 = 13,056,000
  float* p1  = p0 + 13056000;            // 8*3*40*40*85 =  3,264,000
  float* p2  = p1 + 3264000;             // 8*3*20*20*85 =    816,000
  float* obj = p2 + 816000;              // 64*896       =     57,344
  float* gt  = obj + 57344;              // 64

  head_wmma_kernel<128><<<dim3(400, 8), 256, 0, stream>>>(x0, w0, b0, p0, 6400);
  head_wmma_kernel<256><<<dim3(100, 8), 256, 0, stream>>>(x1, w1, b1, p1, 1600);
  head_wmma_kernel<512><<<dim3(25, 8),  256, 0, stream>>>(x2, w2, b2, p2, 400);
  obj_pool_kernel<<<dim3(64, 3), 256, 0, stream>>>(x0, x1, x2, labels, obj, gt);
}
